// MMDLoss_28037546508513
// MI455X (gfx1250) — matrix operand order repaired and checked by reading, not compiled, not run on hardware
//
#include <hip/hip_runtime.h>
#include <stdint.h>

// Energy-MMD on MI455X (gfx1250, wave32).
// Pass 1: convert clouds to bf16 + row norms (once, into d_ws).
// Pass 2: Gram tiles via v_wmma_f32_16x16x32_bf16, 2 column tiles per iteration
//         so independent work fills the WMMA->VALU hazard windows.

typedef __attribute__((ext_vector_type(16))) __bf16         v16bf;
typedef __attribute__((ext_vector_type(16))) unsigned short v16u;
typedef __attribute__((ext_vector_type(8)))  unsigned short v8u;
typedef __attribute__((ext_vector_type(8)))  float          v8f;
typedef __attribute__((ext_vector_type(4)))  float          v4f;

#define S_PTS 2048              // points per cloud
#define DIMS  64                // dims per point
#define TILES (S_PTS / 16)      // 128 WMMA tiles per matrix edge

static __device__ __forceinline__ unsigned short f32_to_bf16_rne(float f) {
    uint32_t u = __float_as_uint(f);
    u += 0x7FFFu + ((u >> 16) & 1u);
    return (unsigned short)(u >> 16);
}
static __device__ __forceinline__ float bf16_bits_to_f32(unsigned short b) {
    return __uint_as_float(((uint32_t)b) << 16);
}

// ---------------------------------------------------------------------------
// Pass 1: one wave per row. Lane l handles dims {2l, 2l+1}: convert to bf16
// (RNE), store packed u32 (coalesced), accumulate |row|^2 from rounded values.
// ---------------------------------------------------------------------------
__global__ __launch_bounds__(256)
void mmd_prep_kernel(const float* __restrict__ x, const float* __restrict__ y,
                     unsigned short* __restrict__ xbf, unsigned short* __restrict__ ybf,
                     float* __restrict__ nx, float* __restrict__ ny, int nRows) {
    const int lane = threadIdx.x & 31;
    int wid = (blockIdx.x * blockDim.x + threadIdx.x) >> 5;
    if (wid >= 2 * nRows) return;
    const float* src; unsigned short* dst; float* nrm; int row;
    if (wid < nRows) { src = x; dst = xbf; nrm = nx; row = wid; }
    else             { src = y; dst = ybf; nrm = ny; row = wid - nRows; }

    const float2 f2 = ((const float2*)(src + (size_t)row * DIMS))[lane];
    const unsigned short b0 = f32_to_bf16_rne(f2.x);
    const unsigned short b1 = f32_to_bf16_rne(f2.y);
    ((uint32_t*)(dst + (size_t)row * DIMS))[lane] = (uint32_t)b0 | ((uint32_t)b1 << 16);

    const float r0 = bf16_bits_to_f32(b0), r1 = bf16_bits_to_f32(b1);
    float n = r0 * r0 + r1 * r1;
    #pragma unroll
    for (int off = 16; off > 0; off >>= 1) n += __shfl_xor(n, off, 32);
    if (lane == 0) nrm[row] = n;
}

// ---------------------------------------------------------------------------
// Fragment loaders from preconverted bf16 (contiguous k order, same in A and B).
// A 16x32: lane half h holds K {kb+16h .. kb+16h+15} contiguous.
// B 32x16: lane (N=lane&15, half h) holds K {kb+16h .. kb+16h+15} contiguous.
// ---------------------------------------------------------------------------
static __device__ __forceinline__ v16bf load_a_frag_bf(const unsigned short* __restrict__ rowPtr,
                                                       int kb, int h8) {
    v8u lo = *(const v8u*)(rowPtr + kb + 2 * h8);
    v8u hi = *(const v8u*)(rowPtr + kb + 2 * h8 + 8);
    v16u s;
    #pragma unroll
    for (int i = 0; i < 8; ++i) { s[i] = lo[i]; s[i + 8] = hi[i]; }
    return __builtin_bit_cast(v16bf, s);
}
static __device__ __forceinline__ v16bf load_b_frag_bf(const unsigned short* __restrict__ rowPtr,
                                                       int off) {
    v16u s = *(const v16u*)(rowPtr + off);     // 32B: two b128 loads
    return __builtin_bit_cast(v16bf, s);
}

// ---------------------------------------------------------------------------
// Pass 2 (fast path): one wave = 16-row x 256-col strip of one distance matrix.
// Two column tiles per loop iteration (software pipelining for hazard cover).
// ---------------------------------------------------------------------------
__global__ __launch_bounds__(256)
void mmd_main_pre_kernel(const unsigned short* __restrict__ xbf,
                         const unsigned short* __restrict__ ybf,
                         const float* __restrict__ nx, const float* __restrict__ ny,
                         float* __restrict__ out, int nBatches) {
    const int lane = threadIdx.x & 31;
    const int wave = threadIdx.x >> 5;
    const int half = lane >> 4;
    const int h8   = half << 3;
    const int ln15 = lane & 15;

    int bid = blockIdx.x;
    const int p     = bid / (nBatches * TILES);       // 0: xy, 1: xx, 2: yy
    const int rem   = bid % (nBatches * TILES);
    const int batch = rem / TILES;
    const int trow  = rem % TILES;

    const unsigned short* Ab = ((p == 2) ? ybf : xbf) + (size_t)batch * S_PTS * DIMS;
    const unsigned short* Bb = ((p == 1) ? xbf : ybf) + (size_t)batch * S_PTS * DIMS;
    const float* nAarr = ((p == 2) ? ny : nx) + (size_t)batch * S_PTS;
    const float* nBarr = ((p == 1) ? nx : ny) + (size_t)batch * S_PTS;
    const float  w     = (p == 0) ? 1.0f : -0.5f;

    const unsigned short* arow = Ab + (size_t)(trow * 16 + ln15) * DIMS;
    const v16bf a0 = load_a_frag_bf(arow, 0,  h8);
    const v16bf a1 = load_a_frag_bf(arow, 32, h8);
    // norms for rows M = half*8 + 0..7 of this strip: one aligned 32B load
    const v8f nA8 = *(const v8f*)(nAarr + trow * 16 + h8);

    float sum = 0.f;
    #pragma unroll 1
    for (int i = 0; i < TILES / 16; ++i) {            // 8 iters x 2 tiles
        const int c0 = wave * (TILES / 8) + 2 * i;

        const unsigned short* brow0 = Bb + (size_t)(c0 * 16 + ln15) * DIMS;
        const unsigned short* brow1 = brow0 + 16 * DIMS;

        const v16bf b00 = load_b_frag_bf(brow0, (half << 4));
        const v16bf b01 = load_b_frag_bf(brow0, 32 + (half << 4));
        const v16bf b10 = load_b_frag_bf(brow1, (half << 4));
        const v16bf b11 = load_b_frag_bf(brow1, 32 + (half << 4));
        const float nB0 = nBarr[c0 * 16 + ln15];      // norm for N = lane&15
        const float nB1 = nBarr[c0 * 16 + 16 + ln15];

        v8f acc0 = {0.f, 0.f, 0.f, 0.f, 0.f, 0.f, 0.f, 0.f};
        v8f acc1 = {0.f, 0.f, 0.f, 0.f, 0.f, 0.f, 0.f, 0.f};
        acc0 = __builtin_amdgcn_wmma_f32_16x16x32_bf16(false, a0, false, b00,
                                                       (short)0, acc0, false, false);
        acc1 = __builtin_amdgcn_wmma_f32_16x16x32_bf16(false, a0, false, b10,
                                                       (short)0, acc1, false, false);
        acc0 = __builtin_amdgcn_wmma_f32_16x16x32_bf16(false, a1, false, b01,
                                                       (short)0, acc0, false, false);
        acc1 = __builtin_amdgcn_wmma_f32_16x16x32_bf16(false, a1, false, b11,
                                                       (short)0, acc1, false, false);

        #pragma unroll
        for (int v = 0; v < 8; ++v) {
            const float d2a = fmaxf(nA8[v] + nB0 - 2.0f * acc0[v], 0.f);
            const float d2b = fmaxf(nA8[v] + nB1 - 2.0f * acc1[v], 0.f);
            sum += __builtin_amdgcn_sqrtf(d2a);       // raw v_sqrt_f32; sqrt(0)==0
            sum += __builtin_amdgcn_sqrtf(d2b);
        }
    }

    #pragma unroll
    for (int off = 16; off > 0; off >>= 1) sum += __shfl_xor(sum, off, 32);

    __shared__ float red[8];
    if (lane == 0) red[wave] = sum;
    __syncthreads();
    if (threadIdx.x == 0) {
        float t = 0.f;
        #pragma unroll
        for (int k = 0; k < 8; ++k) t += red[k];
        atomicAdd(out, t * (w / ((float)nBatches * (float)S_PTS * (float)S_PTS)));
    }
}

// ---------------------------------------------------------------------------
// Fallback (if d_ws is too small): inline conversion, same lean epilogue.
// ---------------------------------------------------------------------------
static __device__ __forceinline__ v16bf cvt_frag(const v4f q0, const v4f q1,
                                                 const v4f q2, const v4f q3, float& norm2) {
    float f[16] = { q0.x,q0.y,q0.z,q0.w, q1.x,q1.y,q1.z,q1.w,
                    q2.x,q2.y,q2.z,q2.w, q3.x,q3.y,q3.z,q3.w };
    v16u s;
    #pragma unroll
    for (int i = 0; i < 16; ++i) {
        const unsigned short b = f32_to_bf16_rne(f[i]);
        const float r = bf16_bits_to_f32(b);
        norm2 += r * r;
        s[i] = b;
    }
    return __builtin_bit_cast(v16bf, s);
}

__global__ __launch_bounds__(256)
void mmd_main_inline_kernel(const float* __restrict__ x, const float* __restrict__ y,
                            float* __restrict__ out, int nBatches) {
    const int lane = threadIdx.x & 31;
    const int wave = threadIdx.x >> 5;
    const int half = lane >> 4;
    const int h8   = half << 3;
    const int ln15 = lane & 15;

    int bid = blockIdx.x;
    const int p     = bid / (nBatches * TILES);
    const int rem   = bid % (nBatches * TILES);
    const int batch = rem / TILES;
    const int trow  = rem % TILES;

    const float* Abase = ((p == 2) ? y : x) + (size_t)batch * S_PTS * DIMS;
    const float* Bbase = ((p == 1) ? x : y) + (size_t)batch * S_PTS * DIMS;
    const float  w     = (p == 0) ? 1.0f : -0.5f;

    const float* arow = Abase + (size_t)(trow * 16 + ln15) * DIMS;
    float an = 0.f;
    const v16bf a0 = cvt_frag(*(const v4f*)(arow + 2 * h8),      *(const v4f*)(arow + 2 * h8 + 4),
                              *(const v4f*)(arow + 2 * h8 + 8), *(const v4f*)(arow + 2 * h8 + 12), an);
    const v16bf a1 = cvt_frag(*(const v4f*)(arow + 32 + 2 * h8),      *(const v4f*)(arow + 32 + 2 * h8 + 4),
                              *(const v4f*)(arow + 32 + 2 * h8 + 8), *(const v4f*)(arow + 32 + 2 * h8 + 12), an);
    const float anFull = an + __shfl_xor(an, 16, 32);
    float nA8[8];
    #pragma unroll
    for (int v = 0; v < 8; ++v) nA8[v] = __shfl(anFull, v + h8, 32);

    float sum = 0.f;
    #pragma unroll 1
    for (int i = 0; i < TILES / 8; ++i) {
        const int c = wave * (TILES / 8) + i;
        const float* brow = Bbase + (size_t)(c * 16 + ln15) * DIMS;
        float bn = 0.f;
        const float* bp0 = brow + (half << 4);
        const float* bp1 = brow + 32 + (half << 4);
        const v16bf b0 = cvt_frag(*(const v4f*)(bp0), *(const v4f*)(bp0 + 4),
                                  *(const v4f*)(bp0 + 8), *(const v4f*)(bp0 + 12), bn);
        const v16bf b1 = cvt_frag(*(const v4f*)(bp1), *(const v4f*)(bp1 + 4),
                                  *(const v4f*)(bp1 + 8), *(const v4f*)(bp1 + 12), bn);
        const float nB = bn + __shfl_xor(bn, 16, 32);

        v8f acc = {0.f, 0.f, 0.f, 0.f, 0.f, 0.f, 0.f, 0.f};
        acc = __builtin_amdgcn_wmma_f32_16x16x32_bf16(false, a0, false, b0,
                                                      (short)0, acc, false, false);
        acc = __builtin_amdgcn_wmma_f32_16x16x32_bf16(false, a1, false, b1,
                                                      (short)0, acc, false, false);
        #pragma unroll
        for (int v = 0; v < 8; ++v) {
            const float d2 = fmaxf(nA8[v] + nB - 2.0f * acc[v], 0.f);
            sum += __builtin_amdgcn_sqrtf(d2);
        }
    }

    #pragma unroll
    for (int off = 16; off > 0; off >>= 1) sum += __shfl_xor(sum, off, 32);

    __shared__ float red[8];
    if (lane == 0) red[wave] = sum;
    __syncthreads();
    if (threadIdx.x == 0) {
        float t = 0.f;
        #pragma unroll
        for (int k = 0; k < 8; ++k) t += red[k];
        atomicAdd(out, t * (w / ((float)nBatches * (float)S_PTS * (float)S_PTS)));
    }
}

// ---------------------------------------------------------------------------
extern "C" void kernel_launch(void* const* d_in, const int* in_sizes, int n_in,
                              void* d_out, int out_size, void* d_ws, size_t ws_size,
                              hipStream_t stream) {
    const float* x = (const float*)d_in[0];
    const float* y = (const float*)d_in[1];
    float* out = (float*)d_out;

    const int nRows    = in_sizes[0] / DIMS;            // 32768
    const int nBatches = nRows / S_PTS;                 // 16
    const int blocks   = 3 * nBatches * TILES;          // 6144

    hipMemsetAsync(out, 0, sizeof(float), stream);

    const size_t bfBytes = (size_t)nRows * DIMS * sizeof(unsigned short); // 4 MB each
    const size_t nmBytes = (size_t)nRows * sizeof(float);                 // 128 KB each
    const size_t need = 2 * bfBytes + 2 * nmBytes;

    if (ws_size >= need) {
        unsigned short* xbf = (unsigned short*)d_ws;
        unsigned short* ybf = (unsigned short*)((char*)d_ws + bfBytes);
        float* nx = (float*)((char*)d_ws + 2 * bfBytes);
        float* ny = (float*)((char*)d_ws + 2 * bfBytes + nmBytes);

        const int prepBlocks = (2 * nRows + 7) / 8;     // one wave per row
        mmd_prep_kernel<<<prepBlocks, 256, 0, stream>>>(x, y, xbf, ybf, nx, ny, nRows);
        mmd_main_pre_kernel<<<blocks, 256, 0, stream>>>(xbf, ybf, nx, ny, out, nBatches);
    } else {
        mmd_main_inline_kernel<<<blocks, 256, 0, stream>>>(x, y, out, nBatches);
    }
}
